// LSTM_RNN_81758997446864
// MI455X (gfx1250) — matrix operand order repaired and checked by reading, not compiled, not run on hardware
//
#include <hip/hip_runtime.h>

// Problem constants (match reference)
#define B_   128
#define T_   1024
#define I_   64
#define H_   256
#define FH   1024      // 4*H
#define NWG  16        // persistent workgroups; each owns 16 hidden columns (all 4 gates)
#define NT   4         // N-tiles per WG (one 16-col tile per gate)
#define KT   10        // K-tiles: 320 / 32  (K = I + H = 64 + 256)
#define FRAG_DW (32*8) // dwords per packed B-fragment (32 lanes x 8 dwords)

typedef __attribute__((ext_vector_type(16))) __bf16 v16bf;
typedef __attribute__((ext_vector_type(8)))  float  v8f;
typedef __attribute__((ext_vector_type(8)))  unsigned int v8u;

static __device__ __forceinline__ unsigned short f32_to_bf16(float f) {
  unsigned u = __builtin_bit_cast(unsigned, f);
  unsigned r = (u + 0x7FFFu + ((u >> 16) & 1u)) >> 16;   // round-to-nearest-even
  return (unsigned short)r;
}
static __device__ __forceinline__ float bf16_to_f32(unsigned short h) {
  return __builtin_bit_cast(float, ((unsigned)h) << 16);
}
static __device__ __forceinline__ unsigned pack2bf(float lo, float hi) {
  return (unsigned)f32_to_bf16(lo) | ((unsigned)f32_to_bf16(hi) << 16);
}
static __device__ __forceinline__ float sigm_f(float x) {
  return 1.0f / (1.0f + __expf(-x));
}
static __device__ __forceinline__ float tanh_f(float x) {
  float e = __expf(-2.0f * x);
  return (1.0f - e) / (1.0f + e);
}

// ---------------------------------------------------------------------------
// Kernel 1: convert x [B,T,I] f32 -> bf16 pairs (u32), row-major (I innermost).
// A-matrix K-pairs are then single dword loads.
// ---------------------------------------------------------------------------
__global__ void xconv_kernel(const float* __restrict__ x, unsigned* __restrict__ xbf) {
  long p = (long)blockIdx.x * blockDim.x + threadIdx.x;
  long np = (long)B_ * T_ * I_ / 2;
  if (p < np) xbf[p] = pack2bf(x[2 * p], x[2 * p + 1]);
}

// ---------------------------------------------------------------------------
// Kernel 2: pack W[I,4H] & U[H,4H] (f32, row-major) into bf16 B-matrix
// fragments in WMMA lane layout; also zero h ping-pong buffers + barrier.
//
// B-fragment layout (32x16 bf16, 8 VGPRs/lane), contiguous K order (same as A, not ISA table order):
//   lane L: N = L%16, hi = L/16 ; VGPR r holds K = ktile*32 + hi*16 + 2r (+1)
// Fragment linear order: [wg][nt][kt][lane][r]
// Column for (wg, nt, n): gate nt -> col = nt*H + wg*16 + n
// Row K: K < 64 -> W[K][col] ; else U[K-64][col]
// ---------------------------------------------------------------------------
__global__ void pack_kernel(const float* __restrict__ W, const float* __restrict__ U,
                            unsigned* __restrict__ wpack, unsigned* __restrict__ hzero,
                            unsigned* __restrict__ bar) {
  long idx = (long)blockIdx.x * blockDim.x + threadIdx.x;
  if (idx == 0) *bar = 0u;
  if (idx < (long)(2 * B_ * H_ / 2)) hzero[idx] = 0u;     // zero both h buffers (bf16 pairs)

  long total = (long)NWG * NT * KT * FRAG_DW;
  if (idx >= total) return;
  int r    = (int)(idx & 7);
  int lane = (int)((idx >> 3) & 31);
  int frag = (int)(idx >> 8);                 // wg*NT*KT + nt*KT + kt
  int kt = frag % KT;
  int nt = (frag / KT) % NT;
  int wg = frag / (KT * NT);
  int n  = lane & 15;
  int hi = lane >> 4;
  int col = nt * H_ + wg * 16 + n;
  int k0  = kt * 32 + hi * 16 + r * 2;
  float f0 = (k0 < I_)     ? W[(long)k0 * FH + col]       : U[(long)(k0 - I_) * FH + col];
  float f1 = (k0 + 1 < I_) ? W[(long)(k0 + 1) * FH + col] : U[(long)(k0 + 1 - I_) * FH + col];
  wpack[idx] = pack2bf(f0, f1);
}

// ---------------------------------------------------------------------------
// Device-wide barrier for the 16 persistent workgroups (release/acquire).
// ---------------------------------------------------------------------------
static __device__ __forceinline__ void grid_sync(unsigned* bar, unsigned want) {
  __threadfence();
  __syncthreads();
  if (threadIdx.x == 0) {
    __hip_atomic_fetch_add(bar, 1u, __ATOMIC_RELEASE, __HIP_MEMORY_SCOPE_AGENT);
    while (__hip_atomic_load(bar, __ATOMIC_ACQUIRE, __HIP_MEMORY_SCOPE_AGENT) < want) {
      __builtin_amdgcn_s_sleep(2);
    }
  }
  __syncthreads();
}

// ---------------------------------------------------------------------------
// Kernel 3: persistent LSTM. 16 WGs x 256 threads (8 waves).
// Wave m handles batch M-tile m (rows m*16 .. m*16+15).
// Each step: Z[128 x 64] = [x_t | h] (128x320 bf16) @ Wpack_wg (320x64) via WMMA,
// then gates + cell update in registers, h_new (bf16) -> global ping-pong.
// ---------------------------------------------------------------------------
__global__ void __launch_bounds__(256)
lstm_kernel(const unsigned* __restrict__ xbf,    // [B][T][I/2] bf16 pairs
            const unsigned* __restrict__ wpack,  // packed B fragments
            const float* __restrict__ bvec,      // [4H] bias
            unsigned short* __restrict__ hbuf,   // [2][B][H] bf16 ping-pong
            unsigned* __restrict__ bar) {
  __shared__ unsigned ldsW[NT * KT * FRAG_DW];   // 10240 dwords = 40 KB

  const int wg   = blockIdx.x;
  const int tid  = threadIdx.x;
  const int lane = tid & 31;
  const int wave = tid >> 5;                     // M-tile index 0..7
  const int n    = lane & 15;
  const int hi   = lane >> 4;

  // Stage this WG's packed weights into LDS (read every step via ds_load_b128).
  const unsigned* wsrc = wpack + (long)wg * NT * KT * FRAG_DW;
  for (int i = tid; i < NT * KT * FRAG_DW; i += 256) ldsW[i] = wsrc[i];
  __syncthreads();

  // C/D fragment mapping: lane L -> N = L%16 ; VGPR v -> M = v + 8*(L/16)
  const int j   = wg * 16 + n;                   // this lane's hidden column
  const int bb0 = wave * 16 + 8 * hi;            // batch rows bb0 .. bb0+7 (per VGPR v)
  float bias[4];
#pragma unroll
  for (int g = 0; g < 4; ++g) bias[g] = bvec[g * H_ + j];

  float c[8];
#pragma unroll
  for (int v = 0; v < 8; ++v) c[v] = 0.0f;

  // A fragment mapping: lane L -> M = L%16 ; VGPR r -> K = hi*16 + 2r
  const int arow = wave * 16 + n;                // this lane's batch row for A
  const unsigned* xrow_base = xbf + (long)arow * T_ * (I_ / 2);

  for (int t = 0; t < T_; ++t) {
    const unsigned short* hcur = hbuf + (long)(t & 1) * (B_ * H_);
    unsigned short*       hnxt = hbuf + (long)((t + 1) & 1) * (B_ * H_);
    const unsigned* xrow = xrow_base + (long)t * (I_ / 2);
    const unsigned short* hrow = hcur + (long)arow * H_;

    // ---- load all 10 A fragments for this wave's M-tile ----
    v16bf afrag[KT];
#pragma unroll
    for (int kt = 0; kt < KT; ++kt) {
      v8u raw;
#pragma unroll
      for (int r = 0; r < 8; ++r) {
        int kofs = (hi << 4) + (r << 1);  // even
        if (kt < 2) {
          raw[r] = xrow[(kt * 32 + kofs) >> 1];                   // x part, K<64
        } else {
          raw[r] = *(const unsigned*)(hrow + ((kt - 2) * 32 + kofs)); // h part
        }
      }
      afrag[kt] = __builtin_bit_cast(v16bf, raw);
    }

    if (t + 1 < T_) __builtin_prefetch(xrow + (I_ / 2), 0, 1);    // global_prefetch_b8

    // ---- GEMM: one f32 accumulator per gate ----
    v8f acc[NT];
#pragma unroll
    for (int nt = 0; nt < NT; ++nt) {
      v8f a0 = {0.f, 0.f, 0.f, 0.f, 0.f, 0.f, 0.f, 0.f};
#pragma unroll
      for (int kt = 0; kt < KT; ++kt) {
        const v8u* bp = (const v8u*)&ldsW[((nt * KT + kt) * 32 + lane) * 8];
        v16bf bfrag = __builtin_bit_cast(v16bf, *bp);
        a0 = __builtin_amdgcn_wmma_f32_16x16x32_bf16(
            false, afrag[kt], false, bfrag, (short)0, a0, false, false);
      }
      acc[nt] = a0;
    }

    // ---- gates + cell update (Keras order i, f, g, o) ----
#pragma unroll
    for (int v = 0; v < 8; ++v) {
      float ig = sigm_f(acc[0][v] + bias[0]);
      float fg = sigm_f(acc[1][v] + bias[1]);
      float gg = tanh_f(acc[2][v] + bias[2]);
      float og = sigm_f(acc[3][v] + bias[3]);
      c[v] = fg * c[v] + ig * gg;
      float hv = og * tanh_f(c[v]);
      hnxt[(long)(bb0 + v) * H_ + j] = f32_to_bf16(hv);
    }

    // ---- device-wide step barrier (double-buffered h, so one per step) ----
    grid_sync(bar, (unsigned)(t + 1) * NWG);
  }
}

// ---------------------------------------------------------------------------
// Kernel 4: Dense(1): out[bb] = sum_j h_last[bb,j]*Wd[j] + bd  (deterministic)
// T is even, so h_last lives in ping buffer 0.
// ---------------------------------------------------------------------------
__global__ void dense_kernel(const unsigned short* __restrict__ hlast,
                             const float* __restrict__ Wd, const float* __restrict__ bd,
                             float* __restrict__ out) {
  int bb = blockIdx.x * blockDim.x + threadIdx.x;
  if (bb >= B_) return;
  float s = bd[0];
  for (int jj = 0; jj < H_; ++jj) s += bf16_to_f32(hlast[(long)bb * H_ + jj]) * Wd[jj];
  out[bb] = s;
}

extern "C" void kernel_launch(void* const* d_in, const int* in_sizes, int n_in,
                              void* d_out, int out_size, void* d_ws, size_t ws_size,
                              hipStream_t stream) {
  const float* x  = (const float*)d_in[0];   // [B,T,I]
  const float* W  = (const float*)d_in[1];   // [I,4H]
  const float* U  = (const float*)d_in[2];   // [H,4H]
  const float* bv = (const float*)d_in[3];   // [4H]
  const float* Wd = (const float*)d_in[4];   // [H]
  const float* bd = (const float*)d_in[5];   // [1]
  float* out = (float*)d_out;

  // Workspace layout
  unsigned* xbf = (unsigned*)d_ws;                                   // B*T*I/2 u32 (16 MB)
  unsigned* wpack = xbf + (long)B_ * T_ * I_ / 2;                    // NWG*NT*KT*256 u32
  unsigned short* hbuf = (unsigned short*)(wpack + (long)NWG * NT * KT * FRAG_DW); // 2*B*H bf16
  unsigned* bar = (unsigned*)(hbuf + 2 * B_ * H_);

  long np = (long)B_ * T_ * I_ / 2;
  xconv_kernel<<<(int)((np + 255) / 256), 256, 0, stream>>>(x, xbf);

  long packN = (long)NWG * NT * KT * FRAG_DW;                        // 163840
  pack_kernel<<<(int)((packN + 255) / 256), 256, 0, stream>>>(W, U, wpack,
                                                              (unsigned*)hbuf, bar);

  lstm_kernel<<<NWG, 256, 0, stream>>>(xbf, wpack, bv, hbuf, bar);

  dense_kernel<<<1, 128, 0, stream>>>(hbuf /* ping 0 == h_last */, Wd, bd, out);
}
